// DirectSpec_20349555048597
// MI455X (gfx1250) — compile-verified, matrix operand order checked by reading, not run on hardware
//
#include <hip/hip_runtime.h>
#include <hip/hip_bf16.h>
#include <cstddef>

// ---------------------------------------------------------------------------
// Problem constants (match reference)
// ---------------------------------------------------------------------------
#define BATCH   8192
#define D       256
#define ALPHA   0.8f
#define TAU     3.0f
#define LOG2E   1.4426950408889634f

typedef _Float16 half8 __attribute__((ext_vector_type(8)));
typedef _Float16 v16h  __attribute__((ext_vector_type(16)));
typedef float    v8f   __attribute__((ext_vector_type(8)));

union AFrag { v16h v; half8 h[2]; };

// ---------------------------------------------------------------------------
// Kernel 1: gather rows by index and L2-normalize; emit f32 + f16 copies.
// One block (256 threads) per batch row.
// ---------------------------------------------------------------------------
__global__ __launch_bounds__(256)
void gather_norm_kernel(const float* __restrict__ embed,
                        const int* __restrict__ idx,
                        float* __restrict__ Xf,
                        _Float16* __restrict__ Xh) {
    const int b = blockIdx.x;
    const int t = threadIdx.x;               // 0..255 == feature index
    const int row = idx[b];
    const float v = embed[(size_t)row * D + t];

    // block reduction of v*v (8 waves of 32)
    __shared__ float red[8];
    float s = v * v;
    #pragma unroll
    for (int m = 16; m; m >>= 1) s += __shfl_xor(s, m, 32);
    const int wv = t >> 5, ln = t & 31;
    if (ln == 0) red[wv] = s;
    __syncthreads();
    if (wv == 0) {
        float r = (ln < 8) ? red[ln] : 0.0f;
        #pragma unroll
        for (int m = 4; m; m >>= 1) r += __shfl_xor(r, m, 32);
        if (ln == 0) red[0] = r;
    }
    __syncthreads();

    const float inv = rsqrtf(red[0]);
    const float xn = v * inv;
    Xf[(size_t)b * D + t] = xn;
    Xh[(size_t)b * D + t] = (_Float16)xn;
}

// ---------------------------------------------------------------------------
// Kernel 2: fused spectral smoothing (flash-attention style; rows are unit
// vectors so tau-scaled scores are bounded by 3 -> no online max needed).
//
//   O_i = sum_j exp(tau * <x_i, x_j>) * x_j ,  l_i = sum_j exp(tau * <x_i,x_j>)
//   y_i = x_i - alpha * O_i / l_i ;  out_i = y_i * ||y_i||
//
// Block = 128 threads = 4 waves; each wave owns a 16-row tile (64 rows/block).
// Q fragments are pre-scaled by tau*log2(e) so p = exp2(s) directly; the
// argument is bounded by tau*log2e ~ 4.33, so the raw v_exp_f32 (no subnormal
// range fixup) is exact enough and co-executes as a TRANS op.
// ---------------------------------------------------------------------------
__global__ __launch_bounds__(128, 1)
void spec_smooth_kernel(const float* __restrict__ Xf,
                        const _Float16* __restrict__ Xh,
                        float* __restrict__ Y) {
    // LDS staging. Strides padded so half8 (16B) accesses are conflict-free.
    __shared__ __align__(16) _Float16 Xs [32][264];   // row-major X block
    __shared__ __align__(16) _Float16 XsT[256][40];   // transposed X block
    __shared__ __align__(16) _Float16 Psh[4][16][40]; // per-wave P staging

    const int tid = threadIdx.x;
    const int wv  = tid >> 5;
    const int ln  = tid & 31;
    const int i0  = blockIdx.x * 64 + wv * 16;   // first row of this wave's tile

    const int nlo = ln & 15;          // A-layout: M / B-layout: N index
    const int kb  = (ln >> 4) * 8;    // K sub-base for 16-bit frag layout

    // ---- load Q A-fragments (16 x 256 f16), pre-scaled by tau*log2e ----
    AFrag qa[8];
    {
        const _Float16 qscale = (_Float16)(TAU * LOG2E);
        const _Float16* qrow = Xh + (size_t)(i0 + nlo) * D;
        #pragma unroll
        for (int kc = 0; kc < 8; ++kc) {
            half8 h0 = *(const half8*)(qrow + kc * 32 + kb);
            half8 h1 = *(const half8*)(qrow + kc * 32 + kb + 16);
            #pragma unroll
            for (int e = 0; e < 8; ++e) { h0[e] *= qscale; h1[e] *= qscale; }
            qa[kc].h[0] = h0;
            qa[kc].h[1] = h1;
        }
    }

    // ---- accumulators: O (16x256 f32 as 16 C-tiles) and row-sum l ----
    v8f o[16];
    v8f lacc;
    #pragma unroll
    for (int nt = 0; nt < 16; ++nt)
        #pragma unroll
        for (int e = 0; e < 8; ++e) o[nt][e] = 0.0f;
    #pragma unroll
    for (int e = 0; e < 8; ++e) lacc[e] = 0.0f;

    // ---- stream over all 8192 columns in blocks of 32 ----
    for (int j = 0; j < BATCH; j += 32) {
        // cooperative stage of X[j:j+32][:] into Xs (row-major) + XsT (transposed)
        for (int it = tid; it < 32 * 32; it += 128) {
            const int r  = it >> 5;   // 0..31 row within block
            const int cg = it & 31;   // column group of 8 halfs
            half8 hv = *(const half8*)(Xh + (size_t)(j + r) * D + cg * 8);
            *(half8*)&Xs[r][cg * 8] = hv;
            #pragma unroll
            for (int e = 0; e < 8; ++e) XsT[cg * 8 + e][r] = hv[e];
            // warm next block (L2-resident; lowers to global_prefetch)
            if (j + 32 < BATCH)
                __builtin_prefetch(Xh + (size_t)(j + 32 + r) * D + cg * 8, 0, 1);
        }
        __syncthreads();

        // ---- scores: two 16x16 tiles (columns j..j+15, j+16..j+31) ----
        #pragma unroll
        for (int t2 = 0; t2 < 2; ++t2) {
            // preload all 8 B-fragments -> batched ds_loads, pipelined waits,
            // then a back-to-back WMMA chain
            AFrag bfs[8];
            #pragma unroll
            for (int kc = 0; kc < 8; ++kc) {
                const _Float16* brow = &Xs[t2 * 16 + nlo][kc * 32 + kb];
                bfs[kc].h[0] = *(const half8*)brow;
                bfs[kc].h[1] = *(const half8*)(brow + 16);
            }
            v8f s;
            #pragma unroll
            for (int e = 0; e < 8; ++e) s[e] = 0.0f;
            #pragma unroll
            for (int kc = 0; kc < 8; ++kc)
                s = __builtin_amdgcn_wmma_f32_16x16x32_f16(
                        false, qa[kc].v, false, bfs[kc].v, (short)0, s, false, false);

            // p = exp2(s) with raw v_exp_f32 (argument bounded, no fixup needed)
            #pragma unroll
            for (int r = 0; r < 8; ++r) {
                const float p = __builtin_amdgcn_exp2f(s[r]);
                lacc[r] += p;
                // stage P in (m,k) row-major form for re-load as A-frag
                Psh[wv][r + 8 * (ln >> 4)][t2 * 16 + nlo] = (_Float16)p;
            }
        }

        // ---- read P back as one 16x32 A-fragment (per-wave, no barrier) ----
        AFrag pa;
        pa.h[0] = *(const half8*)&Psh[wv][nlo][kb];
        pa.h[1] = *(const half8*)&Psh[wv][nlo][kb + 16];

        // ---- O += P @ X_block : 16 N-tiles, software-pipelined in groups of 4
        #pragma unroll
        for (int g = 0; g < 4; ++g) {
            AFrag bo[4];
            #pragma unroll
            for (int q = 0; q < 4; ++q) {
                const _Float16* brow = &XsT[(g * 4 + q) * 16 + nlo][kb];
                bo[q].h[0] = *(const half8*)brow;
                bo[q].h[1] = *(const half8*)(brow + 16);
            }
            #pragma unroll
            for (int q = 0; q < 4; ++q) {
                const int nt = g * 4 + q;
                o[nt] = __builtin_amdgcn_wmma_f32_16x16x32_f16(
                            false, pa.v, false, bo[q].v, (short)0, o[nt], false, false);
            }
        }
        __syncthreads();   // before next block overwrites Xs/XsT
    }

    // ---- epilogue: l row-reduce, y = x - alpha*O/l, norm rescale, store ----
    #pragma unroll
    for (int r = 0; r < 8; ++r) {
        float l = lacc[r];
        #pragma unroll
        for (int m = 8; m; m >>= 1) l += __shfl_xor(l, m, 16);
        lacc[r] = l;
    }

    const int mhi = 8 * (ln >> 4);
    v8f nacc;
    #pragma unroll
    for (int e = 0; e < 8; ++e) nacc[e] = 0.0f;

    #pragma unroll
    for (int nt = 0; nt < 16; ++nt) {
        #pragma unroll
        for (int r = 0; r < 8; ++r) {
            const int row = i0 + mhi + r;
            const int c   = nt * 16 + nlo;
            const float x = Xf[(size_t)row * D + c];
            const float y = x - ALPHA * (o[nt][r] / lacc[r]);
            o[nt][r] = y;
            nacc[r] += y * y;
        }
    }
    #pragma unroll
    for (int r = 0; r < 8; ++r) {
        float n2 = nacc[r];
        #pragma unroll
        for (int m = 8; m; m >>= 1) n2 += __shfl_xor(n2, m, 16);
        nacc[r] = sqrtf(n2);   // SHRINK_NORM == 1.0 -> scale y by its norm
    }
    #pragma unroll
    for (int nt = 0; nt < 16; ++nt) {
        #pragma unroll
        for (int r = 0; r < 8; ++r) {
            const int row = i0 + mhi + r;
            const int c   = nt * 16 + nlo;
            Y[(size_t)row * D + c] = o[nt][r] * nacc[r];
        }
    }
}

// ---------------------------------------------------------------------------
// Kernel 3: loss = mean_b softplus(-<yu_b, yp_b>)  (== -mean log sigmoid(dot))
// ---------------------------------------------------------------------------
__global__ void zero_out_kernel(float* out) { out[0] = 0.0f; }

__global__ __launch_bounds__(256)
void loss_kernel(const float* __restrict__ Yu,
                 const float* __restrict__ Yp,
                 float* __restrict__ out) {
    const int b = blockIdx.x;
    const int t = threadIdx.x;
    float d = Yu[(size_t)b * D + t] * Yp[(size_t)b * D + t];
    #pragma unroll
    for (int m = 16; m; m >>= 1) d += __shfl_xor(d, m, 32);
    __shared__ float red[8];
    if ((t & 31) == 0) red[t >> 5] = d;
    __syncthreads();
    if (t < 32) {
        float r = (t < 8) ? red[t] : 0.0f;
        #pragma unroll
        for (int m = 4; m; m >>= 1) r += __shfl_xor(r, m, 32);
        if (t == 0) {
            const float dd = r;
            // stable softplus(-dd)
            const float sp = fmaxf(-dd, 0.0f) + log1pf(__expf(-fabsf(dd)));
            atomicAdd(out, sp / (float)BATCH);
        }
    }
}

// ---------------------------------------------------------------------------
// Launch
// ---------------------------------------------------------------------------
extern "C" void kernel_launch(void* const* d_in, const int* in_sizes, int n_in,
                              void* d_out, int out_size, void* d_ws, size_t ws_size,
                              hipStream_t stream) {
    const float* user_embed = (const float*)d_in[0];
    const float* item_embed = (const float*)d_in[1];
    const int*   u          = (const int*)d_in[2];
    const int*   p          = (const int*)d_in[3];
    float*       out        = (float*)d_out;

    // workspace layout
    const size_t nE = (size_t)BATCH * D;
    char* ws = (char*)d_ws;
    float*    Xuf = (float*)ws;                 ws += nE * sizeof(float);
    float*    Xpf = (float*)ws;                 ws += nE * sizeof(float);
    float*    Yu  = (float*)ws;                 ws += nE * sizeof(float);
    float*    Yp  = (float*)ws;                 ws += nE * sizeof(float);
    _Float16* Xuh = (_Float16*)ws;              ws += nE * sizeof(_Float16);
    _Float16* Xph = (_Float16*)ws;              ws += nE * sizeof(_Float16);

    gather_norm_kernel<<<BATCH, 256, 0, stream>>>(user_embed, u, Xuf, Xuh);
    gather_norm_kernel<<<BATCH, 256, 0, stream>>>(item_embed, p, Xpf, Xph);

    // 64 rows per block -> 128 blocks per matrix
    spec_smooth_kernel<<<BATCH / 64, 128, 0, stream>>>(Xuf, Xuh, Yu);
    spec_smooth_kernel<<<BATCH / 64, 128, 0, stream>>>(Xpf, Xph, Yp);

    zero_out_kernel<<<1, 1, 0, stream>>>(out);
    loss_kernel<<<BATCH, 256, 0, stream>>>(Yu, Yp, out);
}
